// TransMIL_13846974562422
// MI455X (gfx1250) — matrix operand order repaired and checked by reading, not compiled, not run on hardware
//
#include <hip/hip_runtime.h>

// ---------------------------------------------------------------------------
// TransMIL forward for MI455X (gfx1250, wave32, WMMA).
// All dense math runs through v_wmma_f32_16x16x32_f16 (f16 operands, f32 acc).
// GEMM: 128x64x32 tiles, 8 waves x (32x32 sub-tile) = 4 WMMA per K-step,
// 128-bit global->LDS->VGPR movement, no divergence (dims are tile-exact).
// transB is a compile-time template parameter -> straight-line K-loop.
// ---------------------------------------------------------------------------

typedef _Float16 v16h __attribute__((ext_vector_type(16)));
typedef _Float16 v8h  __attribute__((ext_vector_type(8)));
typedef float    v8f  __attribute__((ext_vector_type(8)));

#define TPB 256

__device__ __forceinline__ v16h cat8(v8h a, v8h b) {
  return __builtin_shufflevector(a, b, 0, 1, 2, 3, 4, 5, 6, 7, 8, 9, 10, 11, 12, 13, 14, 15);
}

// ------------------------- elementwise helpers -----------------------------

__global__ void k_assemble(const float* __restrict__ h, const float* __restrict__ cls,
                           float* __restrict__ x, long long n) {
  long long i = (long long)blockIdx.x * blockDim.x + threadIdx.x;
  if (i >= n) return;
  long long row = i >> 9; int col = (int)(i & 511);
  x[i] = (row == 0) ? cls[col] : h[((row - 1) << 9) + col];
}

__global__ void k_layernorm(const float* __restrict__ x, const float* __restrict__ g,
                            const float* __restrict__ b, float* __restrict__ o, int C) {
  long long row = blockIdx.x;
  const float* r = x + row * C;
  __shared__ float s1[TPB], s2[TPB];
  int tid = threadIdx.x;
  float s = 0.f, q = 0.f;
  for (int i = tid; i < C; i += TPB) { float v = r[i]; s += v; q += v * v; }
  s1[tid] = s; s2[tid] = q; __syncthreads();
  for (int st = TPB / 2; st > 0; st >>= 1) {
    if (tid < st) { s1[tid] += s1[tid + st]; s2[tid] += s2[tid + st]; }
    __syncthreads();
  }
  float mu = s1[0] / C, var = s2[0] / C - mu * mu;
  float inv = rsqrtf(var + 1e-5f);
  for (int i = tid; i < C; i += TPB) o[row * C + i] = (r[i] - mu) * inv * g[i] + b[i];
}

__global__ void k_pad_cvt(const float* __restrict__ src, _Float16* __restrict__ dst,
                          int padRows, long long n) {
  long long i = (long long)blockIdx.x * blockDim.x + threadIdx.x;
  if (i >= n) return;
  long long row = i >> 9; int col = (int)(i & 511);
  dst[i] = (row < padRows) ? (_Float16)0.f : (_Float16)src[((row - padRows) << 9) + col];
}

__global__ void k_cvt(const float* __restrict__ a, _Float16* __restrict__ o, long long n) {
  long long i = (long long)blockIdx.x * blockDim.x + threadIdx.x;
  if (i < n) o[i] = (_Float16)a[i];
}

__global__ void k_scale_q(float* __restrict__ p, long long n) {  // scale q cols (ld=1536)
  long long i = (long long)blockIdx.x * blockDim.x + threadIdx.x;
  if (i >= n) return;
  long long r = i >> 9; int c = (int)(i & 511);
  p[r * 1536 + c] *= 0.125f;  // dh^-0.5, dh = 64
}

__global__ void k_add(float* __restrict__ dst, const float* __restrict__ src, long long n) {
  long long i = (long long)blockIdx.x * blockDim.x + threadIdx.x;
  if (i < n) dst[i] += src[i];
}

__global__ void k_copy(float* __restrict__ dst, const float* __restrict__ src, long long n) {
  long long i = (long long)blockIdx.x * blockDim.x + threadIdx.x;
  if (i < n) dst[i] = src[i];
}

// ------------------- generic batched WMMA GEMM (f16 -> f32) -----------------
// C[b] = alpha * A[b] @ op(B[b]) + bias ; op(B)=B (KxN) or B^T (NxK, TRANSB=1)
// REQUIRES: M % 128 == 0, N % 64 == 0, K % 32 == 0 (true for all call sites),
// and 16-byte alignment of A/B rows (true: all lda/ldb are multiples of 8).

template <int TRANSB>
__global__ __launch_bounds__(256) void k_gemm_f16(
    const _Float16* __restrict__ A, const _Float16* __restrict__ B, float* __restrict__ C,
    int M, int N, int K, int lda, int ldb, int ldc,
    long long sA, long long sB, long long sC, float alpha, const float* __restrict__ bias) {
  __shared__ alignas(16) _Float16 As[128 * 40];   // 128 rows x 32 K (stride 40)
  __shared__ alignas(16) _Float16 Bsm[64 * 40];   // TRANSB=0: [32][72]; TRANSB=1: [64][40]
  (void)M;
  int tid = threadIdx.x;
  int wid = tid >> 5, lane = tid & 31;
  int ml = lane & 15, hi = lane >> 4;
  int hi8 = hi * 8, hi16 = hi * 16;
  int m0 = blockIdx.y * 128, n0 = blockIdx.x * 64;
  const _Float16* Ab = A + sA * blockIdx.z;
  const _Float16* Bb = B + sB * blockIdx.z;
  float* Cb = C + sC * blockIdx.z;
  int wm = (wid & 3) * 32;        // 4 wave-rows of 32
  int wn = (wid >> 2) * 32;       // 2 wave-cols of 32
  v8f acc00 = {}, acc01 = {}, acc10 = {}, acc11 = {};

  // cooperative load indices (all 128-bit transfers)
  int ar = tid >> 2;              // 0..63 (and +64)
  int ac = (tid & 3) * 8;         // 0,8,16,24
  int br0 = tid >> 3;             // 0..31  (TRANSB=0)
  int bc0 = (tid & 7) * 8;        // 0..56
  const _Float16* Abase0 = Ab + (long long)(m0 + ar) * lda + ac;
  const _Float16* Abase1 = Ab + (long long)(m0 + ar + 64) * lda + ac;
  const _Float16* Bbase = TRANSB
      ? Bb + (long long)(n0 + ar) * ldb + ac          // n-major, K-contiguous
      : Bb + (long long)br0 * ldb + n0 + bc0;         // k-major, N-contiguous
  long long bstep = TRANSB ? 32 : (long long)32 * ldb;

  for (int k0 = 0; k0 < K; k0 += 32) {
    // ---- A tile: 128x32 halves ----
    *(v8h*)&As[ar * 40 + ac]        = *(const v8h*)(Abase0 + k0);
    *(v8h*)&As[(ar + 64) * 40 + ac] = *(const v8h*)(Abase1 + k0);
    // ---- B tile ----
    if (TRANSB) {
      *(v8h*)&Bsm[ar * 40 + ac] = *(const v8h*)Bbase;
    } else {
      *(v8h*)&Bsm[br0 * 72 + bc0] = *(const v8h*)Bbase;
    }
    // prefetch next K-step tiles into cache while WMMAs run
    if (k0 + 32 < K) {
      __builtin_prefetch(Abase0 + k0 + 32, 0, 3);
      __builtin_prefetch(Abase1 + k0 + 32, 0, 3);
      __builtin_prefetch(Bbase + bstep, 0, 3);
    }
    Bbase += bstep;
    __syncthreads();

    // ---- A fragments (16x32: lanes 0-15 K0..15,
    //      lanes 16-31 K16..31) ----
    int rowA0 = (wm + ml) * 40;
    int rowA1 = (wm + 16 + ml) * 40;
    v16h a0 = cat8(*(const v8h*)&As[rowA0 + hi16], *(const v8h*)&As[rowA0 + hi16 + 8]);
    v16h a1 = cat8(*(const v8h*)&As[rowA1 + hi16], *(const v8h*)&As[rowA1 + hi16 + 8]);

    // ---- B fragments (32x16: lanes 0-15 K0..15, lanes 16-31 K16..31) ----
    v16h b0, b1;
    if (TRANSB) {
      int rowB0 = (wn + ml) * 40;
      int rowB1 = (wn + 16 + ml) * 40;
      b0 = cat8(*(const v8h*)&Bsm[rowB0 + hi16], *(const v8h*)&Bsm[rowB0 + hi16 + 8]);
      b1 = cat8(*(const v8h*)&Bsm[rowB1 + hi16], *(const v8h*)&Bsm[rowB1 + hi16 + 8]);
    } else {
#pragma unroll
      for (int j = 0; j < 16; ++j) {
        b0[j] = Bsm[(hi16 + j) * 72 + wn + ml];
        b1[j] = Bsm[(hi16 + j) * 72 + wn + 16 + ml];
      }
    }

    acc00 = __builtin_amdgcn_wmma_f32_16x16x32_f16(false, a0, false, b0, (short)0, acc00, false, false);
    acc01 = __builtin_amdgcn_wmma_f32_16x16x32_f16(false, a0, false, b1, (short)0, acc01, false, false);
    acc10 = __builtin_amdgcn_wmma_f32_16x16x32_f16(false, a1, false, b0, (short)0, acc10, false, false);
    acc11 = __builtin_amdgcn_wmma_f32_16x16x32_f16(false, a1, false, b1, (short)0, acc11, false, false);
    __syncthreads();
  }

  // ---- store (C layout: VGPR i holds row hi*8+i, col = lane&15) ----
  int c0 = n0 + wn + ml, c1 = c0 + 16;
  float bv0 = bias ? bias[c0] : 0.f;
  float bv1 = bias ? bias[c1] : 0.f;
#pragma unroll
  for (int i = 0; i < 8; ++i) {
    int r0 = m0 + wm + hi8 + i;
    int r1 = r0 + 16;
    Cb[(long long)r0 * ldc + c0] = alpha * acc00[i] + bv0;
    Cb[(long long)r0 * ldc + c1] = alpha * acc01[i] + bv1;
    Cb[(long long)r1 * ldc + c0] = alpha * acc10[i] + bv0;
    Cb[(long long)r1 * ldc + c1] = alpha * acc11[i] + bv1;
  }
}

// -------------------------- softmax / landmarks -----------------------------

__global__ void k_softmax(const float* __restrict__ in, float* __restrict__ outf,
                          _Float16* __restrict__ outh, int C) {
  long long row = blockIdx.x;
  const float* r = in + row * (long long)C;
  __shared__ float sd[TPB];
  int tid = threadIdx.x;
  float mx = -3.4e38f;
  for (int i = tid; i < C; i += TPB) mx = fmaxf(mx, r[i]);
  sd[tid] = mx; __syncthreads();
  for (int s = TPB / 2; s > 0; s >>= 1) { if (tid < s) sd[tid] = fmaxf(sd[tid], sd[tid + s]); __syncthreads(); }
  mx = sd[0]; __syncthreads();
  float sum = 0.f;
  for (int i = tid; i < C; i += TPB) sum += __expf(r[i] - mx);
  sd[tid] = sum; __syncthreads();
  for (int s = TPB / 2; s > 0; s >>= 1) { if (tid < s) sd[tid] += sd[tid + s]; __syncthreads(); }
  float inv = 1.0f / sd[0];
  for (int i = tid; i < C; i += TPB) {
    float p = __expf(r[i] - mx) * inv;
    if (outf) outf[row * (long long)C + i] = p;
    if (outh) outh[row * (long long)C + i] = (_Float16)p;
  }
}

__global__ void k_landmark(const float* __restrict__ src, float* __restrict__ dst) {
  // grid 2048 = (head, landmark); 64 threads = d; mean over 65 consecutive tokens
  int hi = blockIdx.x, d = threadIdx.x;
  int hh = hi >> 8, i = hi & 255;
  float s = 0.f;
  for (int j = 0; j < 65; ++j) s += src[(long long)(i * 65 + j) * 1536 + hh * 64 + d];
  dst[(long long)hi * 64 + d] = s * (1.0f / 65.0f);
}

// ------------------------------ pinv helpers --------------------------------

__global__ void k_abs_row(const float* __restrict__ a, float* __restrict__ rs) {
  int row = blockIdx.x * blockDim.x + threadIdx.x;
  if (row < 2048) {
    const float* p = a + (long long)row * 256;
    float s = 0.f;
    for (int j = 0; j < 256; ++j) s += fabsf(p[j]);
    rs[row] = s;
  }
}

__global__ void k_abs_col(const float* __restrict__ a, float* __restrict__ cs) {
  int idx = blockIdx.x * blockDim.x + threadIdx.x;
  if (idx < 2048) {
    int b = idx >> 8, j = idx & 255;
    const float* p = a + ((long long)b << 16) + j;
    float s = 0.f;
    for (int i = 0; i < 256; ++i) s += fabsf(p[(long long)i * 256]);
    cs[idx] = s;
  }
}

__global__ void k_denom(const float* __restrict__ rs, const float* __restrict__ cs,
                        float* __restrict__ d) {
  __shared__ float s1[TPB], s2[TPB];
  int tid = threadIdx.x;
  float m1 = -1e30f, m2 = -1e30f;
  for (int i = tid; i < 2048; i += TPB) { m1 = fmaxf(m1, rs[i]); m2 = fmaxf(m2, cs[i]); }
  s1[tid] = m1; s2[tid] = m2; __syncthreads();
  for (int st = TPB / 2; st > 0; st >>= 1) {
    if (tid < st) { s1[tid] = fmaxf(s1[tid], s1[tid + st]); s2[tid] = fmaxf(s2[tid], s2[tid + st]); }
    __syncthreads();
  }
  if (tid == 0) d[0] = s1[0] * s2[0];
}

__global__ void k_zinit(const float* __restrict__ a2p, const float* __restrict__ d,
                        _Float16* __restrict__ zh) {
  long long idx = (long long)blockIdx.x * blockDim.x + threadIdx.x;
  if (idx >= (long long)8 * 65536) return;
  int b = (int)(idx >> 16), r = (int)((idx >> 8) & 255), c = (int)(idx & 255);
  zh[idx] = (_Float16)(a2p[((long long)b << 16) + ((long long)c << 8) + r] / d[0]);
}

__global__ void k_idminus(const float* __restrict__ in, _Float16* __restrict__ o, float cval) {
  long long idx = (long long)blockIdx.x * blockDim.x + threadIdx.x;
  if (idx >= (long long)8 * 65536) return;
  int r = (int)((idx >> 8) & 255), c = (int)(idx & 255);
  o[idx] = (_Float16)(((r == c) ? cval : 0.f) - in[idx]);
}

// ------------------------------ conv kernels --------------------------------

__global__ void k_resconv(const float* __restrict__ v, const float* __restrict__ w,
                          float* __restrict__ om) {
  // depthwise 33-tap conv along sequence, per head; v stride 1536, out stride 512
  int i = blockIdx.x, ch = threadIdx.x;
  int hh = ch >> 6;
  float acc = 0.f;
  for (int j = 0; j < 33; ++j) {
    int sidx = i + j - 16;
    if (sidx >= 0 && sidx < 16640) acc += w[hh * 33 + j] * v[(long long)sidx * 1536 + ch];
  }
  om[(long long)i * 512 + ch] += acc;
}

__global__ void k_ppeg(const float* __restrict__ x,
                       const float* __restrict__ w7, const float* __restrict__ b7,
                       const float* __restrict__ w5, const float* __restrict__ b5,
                       const float* __restrict__ w3, const float* __restrict__ b3,
                       float* __restrict__ y) {
  int t = blockIdx.x;          // 0..16383 spatial
  int c = threadIdx.x;         // 0..511 channel
  int r = t >> 7, q = t & 127;
  const float* feat = x + 512; // skip cls row
  float acc = feat[(long long)t * 512 + c] + b7[c] + b5[c] + b3[c];
  for (int dr = -3; dr <= 3; ++dr)
    for (int dc = -3; dc <= 3; ++dc) {
      int rr = r + dr, cc = q + dc;
      if (rr >= 0 && rr < 128 && cc >= 0 && cc < 128)
        acc += w7[c * 49 + (dr + 3) * 7 + (dc + 3)] * feat[(long long)(rr * 128 + cc) * 512 + c];
    }
  for (int dr = -2; dr <= 2; ++dr)
    for (int dc = -2; dc <= 2; ++dc) {
      int rr = r + dr, cc = q + dc;
      if (rr >= 0 && rr < 128 && cc >= 0 && cc < 128)
        acc += w5[c * 25 + (dr + 2) * 5 + (dc + 2)] * feat[(long long)(rr * 128 + cc) * 512 + c];
    }
  for (int dr = -1; dr <= 1; ++dr)
    for (int dc = -1; dc <= 1; ++dc) {
      int rr = r + dr, cc = q + dc;
      if (rr >= 0 && rr < 128 && cc >= 0 && cc < 128)
        acc += w3[c * 9 + (dr + 1) * 3 + (dc + 1)] * feat[(long long)(rr * 128 + cc) * 512 + c];
    }
  y[(long long)t * 512 + c] = acc;
}

// ------------------------------ final head ----------------------------------

__global__ void k_final(const float* __restrict__ x, const float* __restrict__ g,
                        const float* __restrict__ b, const float* __restrict__ cw,
                        const float* __restrict__ cb, float* __restrict__ out) {
  __shared__ float feat[512];
  __shared__ float s1[TPB], s2[TPB];
  int tid = threadIdx.x;
  float s = 0.f, q = 0.f;
  for (int i = tid; i < 512; i += TPB) { float v = x[i]; s += v; q += v * v; }
  s1[tid] = s; s2[tid] = q; __syncthreads();
  for (int st = TPB / 2; st > 0; st >>= 1) {
    if (tid < st) { s1[tid] += s1[tid + st]; s2[tid] += s2[tid + st]; }
    __syncthreads();
  }
  float mu = s1[0] / 512.f, var = s2[0] / 512.f - mu * mu;
  float inv = rsqrtf(var + 1e-5f);
  for (int i = tid; i < 512; i += TPB) {
    float v = (x[i] - mu) * inv * g[i] + b[i];
    v = fmaxf(v, 0.f);
    feat[i] = v;
    out[4 + i] = v;
  }
  __syncthreads();
  if (tid < 4) {
    float acc = cb[tid];
    for (int c = 0; c < 512; ++c) acc += feat[c] * cw[c * 4 + tid];
    out[tid] = acc;
  }
}

// ------------------------------- host side ----------------------------------

extern "C" void kernel_launch(void* const* d_in, const int* in_sizes, int n_in,
                              void* d_out, int out_size, void* d_ws, size_t ws_size,
                              hipStream_t stream) {
  (void)in_sizes; (void)n_in; (void)out_size; (void)ws_size;
  const float* h      = (const float*)d_in[0];
  const float* cls_t  = (const float*)d_in[1];
  const float* ln1_g  = (const float*)d_in[2];
  const float* ln1_b  = (const float*)d_in[3];
  const float* qkv1_w = (const float*)d_in[4];
  const float* out1_w = (const float*)d_in[5];
  const float* out1_b = (const float*)d_in[6];
  const float* res1_w = (const float*)d_in[7];
  const float* ln2_g  = (const float*)d_in[8];
  const float* ln2_b  = (const float*)d_in[9];
  const float* qkv2_w = (const float*)d_in[10];
  const float* out2_w = (const float*)d_in[11];
  const float* out2_b = (const float*)d_in[12];
  const float* res2_w = (const float*)d_in[13];
  const float* p7_w = (const float*)d_in[14]; const float* p7_b = (const float*)d_in[15];
  const float* p5_w = (const float*)d_in[16]; const float* p5_b = (const float*)d_in[17];
  const float* p3_w = (const float*)d_in[18]; const float* p3_b = (const float*)d_in[19];
  const float* nf_g = (const float*)d_in[20]; const float* nf_b = (const float*)d_in[21];
  const float* cls_w = (const float*)d_in[22]; const float* cls_b = (const float*)d_in[23];
  float* out = (float*)d_out;

  const int NT = 16385;   // tokens incl cls (128*128 + 1, add = 0)
  const int NP = 16640;   // padded to multiple of 256 (pad = 255 at front)
  const int L = 512, NQ = 1536;

  // bump allocator over workspace
  size_t off = 0;
  char* bse = (char*)d_ws;
  auto alloc = [&](size_t bytes) -> void* {
    void* p = bse + off;
    off = (off + bytes + 255) & ~(size_t)255;
    return p;
  };

  float*    xcur  = (float*)alloc((size_t)NT * L * 4);
  float*    lnbuf = (float*)alloc((size_t)NT * L * 4);   // also PPEG temp
  _Float16* xph   = (_Float16*)alloc((size_t)NP * L * 2);
  float*    qkv   = (float*)alloc((size_t)NP * NQ * 4);
  _Float16* qkvh  = (_Float16*)alloc((size_t)NP * NQ * 2);
  _Float16* wbf   = (_Float16*)alloc((size_t)L * NQ * 2);
  float*    ql    = (float*)alloc((size_t)8 * 256 * 64 * 4);
  float*    kl    = (float*)alloc((size_t)8 * 256 * 64 * 4);
  _Float16* qlh   = (_Float16*)alloc((size_t)8 * 256 * 64 * 2);
  _Float16* klh   = (_Float16*)alloc((size_t)8 * 256 * 64 * 2);
  float*    scores= (float*)alloc((size_t)8 * NP * 256 * 4);
  _Float16* probs = (_Float16*)alloc((size_t)8 * NP * 256 * 2);
  float*    a2s   = (float*)alloc((size_t)8 * 65536 * 4);
  float*    a2p   = (float*)alloc((size_t)8 * 65536 * 4);
  _Float16* a2ph  = (_Float16*)alloc((size_t)8 * 65536 * 2);
  float*    rs    = (float*)alloc(2048 * 4);
  float*    cs    = (float*)alloc(2048 * 4);
  float*    den   = (float*)alloc(256);
  _Float16* zh    = (_Float16*)alloc((size_t)8 * 65536 * 2);
  float*    xzf   = (float*)alloc((size_t)8 * 65536 * 4);
  _Float16* xzh   = (_Float16*)alloc((size_t)8 * 65536 * 2);
  float*    ptf   = (float*)alloc((size_t)8 * 65536 * 4);
  _Float16* tth   = (_Float16*)alloc((size_t)8 * 65536 * 2);
  float*    pzf   = (float*)alloc((size_t)8 * 65536 * 4);
  float*    kvf   = (float*)alloc((size_t)8 * 256 * 64 * 4);
  _Float16* kvh   = (_Float16*)alloc((size_t)8 * 256 * 64 * 2);
  float*    t1f   = (float*)alloc((size_t)8 * 256 * 64 * 4);
  _Float16* t1h   = (_Float16*)alloc((size_t)8 * 256 * 64 * 2);
  float*    om    = (float*)alloc((size_t)NP * L * 4);
  _Float16* omh   = (_Float16*)alloc((size_t)NP * L * 2);
  float*    aout  = (float*)alloc((size_t)NP * L * 4);

  auto blocks = [](long long n) { return (unsigned)((n + TPB - 1) / TPB); };
  auto cvt = [&](const float* a, _Float16* o, long long n) {
    k_cvt<<<blocks(n), TPB, 0, stream>>>(a, o, n);
  };
  auto gemm = [&](const _Float16* A, const _Float16* B, float* C, int M, int N, int K,
                  int lda, int ldb, int ldc, int tB,
                  long long sA, long long sB, long long sC, int batch,
                  float alpha, const float* bias) {
    dim3 g((N + 63) / 64, (M + 127) / 128, batch);
    if (tB)
      k_gemm_f16<1><<<g, 256, 0, stream>>>(A, B, C, M, N, K, lda, ldb, ldc, sA, sB, sC, alpha, bias);
    else
      k_gemm_f16<0><<<g, 256, 0, stream>>>(A, B, C, M, N, K, lda, ldb, ldc, sA, sB, sC, alpha, bias);
  };

  auto run_block = [&](const float* lg, const float* lb, const float* qw,
                       const float* ow, const float* ob, const float* rw) {
    // x -> LN -> front-pad(255 zero rows) + f16
    k_layernorm<<<NT, TPB, 0, stream>>>(xcur, lg, lb, lnbuf, L);
    k_pad_cvt<<<blocks((long long)NP * L), TPB, 0, stream>>>(lnbuf, xph, 255, (long long)NP * L);
    // QKV projection: [16640,512] @ [512,1536]
    cvt(qw, wbf, (long long)L * NQ);
    gemm(xph, wbf, qkv, NP, NQ, L, L, NQ, NQ, 0, 0, 0, 0, 1, 1.f, nullptr);
    // scale q by dh^-0.5, landmarks (mean over 65-token chunks)
    k_scale_q<<<blocks((long long)NP * L), TPB, 0, stream>>>(qkv, (long long)NP * L);
    k_landmark<<<2048, 64, 0, stream>>>(qkv, ql);
    k_landmark<<<2048, 64, 0, stream>>>(qkv + 512, kl);
    cvt(qkv, qkvh, (long long)NP * NQ);
    cvt(ql, qlh, 8 * 256 * 64);
    cvt(kl, klh, 8 * 256 * 64);
    // attn2 = softmax(q_l @ k_l^T)  [8,256,256]
    gemm(qlh, klh, a2s, 256, 256, 64, 64, 64, 256, 1, 256 * 64, 256 * 64, 65536, 8, 1.f, nullptr);
    k_softmax<<<2048, TPB, 0, stream>>>(a2s, a2p, a2ph, 256);
    // pinv init: z = attn2^T / (max rowsum * max colsum)  (global max)
    k_abs_row<<<8, TPB, 0, stream>>>(a2p, rs);
    k_abs_col<<<8, TPB, 0, stream>>>(a2p, cs);
    k_denom<<<1, TPB, 0, stream>>>(rs, cs, den);
    k_zinit<<<blocks((long long)8 * 65536), TPB, 0, stream>>>(a2p, den, zh);
    // 6 Newton-Schulz iterations: z = 0.25 z (13I - xz(15I - xz(7I - xz)))
    for (int it = 0; it < 6; ++it) {
      gemm(a2ph, zh, xzf, 256, 256, 256, 256, 256, 256, 0, 65536, 65536, 65536, 8, 1.f, nullptr);
      cvt(xzf, xzh, (long long)8 * 65536);
      k_idminus<<<blocks((long long)8 * 65536), TPB, 0, stream>>>(xzf, tth, 7.f);
      gemm(xzh, tth, ptf, 256, 256, 256, 256, 256, 256, 0, 65536, 65536, 65536, 8, 1.f, nullptr);
      k_idminus<<<blocks((long long)8 * 65536), TPB, 0, stream>>>(ptf, tth, 15.f);
      gemm(xzh, tth, ptf, 256, 256, 256, 256, 256, 256, 0, 65536, 65536, 65536, 8, 1.f, nullptr);
      k_idminus<<<blocks((long long)8 * 65536), TPB, 0, stream>>>(ptf, tth, 13.f);
      gemm(zh, tth, pzf, 256, 256, 256, 256, 256, 256, 0, 65536, 65536, 65536, 8, 0.25f, nullptr);
      cvt(pzf, zh, (long long)8 * 65536);
    }
    // attn3 = softmax(q_l @ k^T)  [8,256,16640] ; kv = attn3 @ v  [8,256,64]
    gemm(qlh, qkvh + 512, scores, 256, NP, 64, 64, NQ, NP, 1,
         256 * 64, 64, (long long)256 * NP, 8, 1.f, nullptr);
    k_softmax<<<2048, TPB, 0, stream>>>(scores, nullptr, probs, NP);
    gemm(probs, qkvh + 1024, kvf, 256, 64, NP, NP, NQ, 64, 0,
         (long long)256 * NP, 64, 256 * 64, 8, 1.f, nullptr);
    cvt(kvf, kvh, 8 * 256 * 64);
    // t1 = pinv(attn2) @ kv  [8,256,64]
    gemm(zh, kvh, t1f, 256, 64, 256, 256, 64, 64, 0, 65536, 256 * 64, 256 * 64, 8, 1.f, nullptr);
    cvt(t1f, t1h, 8 * 256 * 64);
    // attn1 = softmax(q @ k_l^T)  [8,16640,256] ; out = attn1 @ t1 (merged heads)
    gemm(qkvh, klh, scores, NP, 256, 64, NQ, 64, 256, 1,
         64, 256 * 64, (long long)NP * 256, 8, 1.f, nullptr);
    k_softmax<<<8 * NP, TPB, 0, stream>>>(scores, nullptr, probs, 256);
    gemm(probs, t1h, om, NP, 64, 256, 256, 64, 512, 0,
         (long long)NP * 256, 256 * 64, 64, 8, 1.f, nullptr);
    // + depthwise 33-tap residual conv on v
    k_resconv<<<NP, 512, 0, stream>>>(qkv + 1024, rw, om);
    // output projection + residual into x (last 16385 rows)
    cvt(om, omh, (long long)NP * L);
    cvt(ow, wbf, (long long)L * L);
    gemm(omh, wbf, aout, NP, L, L, L, L, L, 0, 0, 0, 0, 1, 1.f, ob);
    k_add<<<blocks((long long)NT * L), TPB, 0, stream>>>(xcur, aout + (long long)255 * 512,
                                                         (long long)NT * L);
  };

  // pipeline
  k_assemble<<<blocks((long long)NT * L), TPB, 0, stream>>>(h, cls_t, xcur, (long long)NT * L);
  run_block(ln1_g, ln1_b, qkv1_w, out1_w, out1_b, res1_w);
  k_ppeg<<<16384, 512, 0, stream>>>(xcur, p7_w, p7_b, p5_w, p5_b, p3_w, p3_b, lnbuf);
  k_copy<<<blocks((long long)16384 * 512), TPB, 0, stream>>>(xcur + 512, lnbuf,
                                                             (long long)16384 * 512);
  run_block(ln2_g, ln2_b, qkv2_w, out2_w, out2_b, res2_w);
  k_final<<<1, TPB, 0, stream>>>(xcur, nf_g, nf_b, cls_w, cls_b, out);
}
